// Block_816043786529
// MI455X (gfx1250) — matrix operand order repaired and checked by reading, not compiled, not run on hardware
//
#include <hip/hip_runtime.h>
#include <hip/hip_bf16.h>

typedef __bf16 bf16;
typedef __attribute__((ext_vector_type(8)))  __bf16 v8bf;
typedef __attribute__((ext_vector_type(16))) __bf16 v16bf;
typedef __attribute__((ext_vector_type(8)))  float  v8f;
typedef __attribute__((ext_vector_type(4)))  unsigned int v4u;
typedef __attribute__((ext_vector_type(8)))  int v8i_;
typedef __attribute__((ext_vector_type(4)))  int v4i_;

enum { Bz = 4, Tt = 2048, Cc = 768, Hh = 12, HDd = 64, C3 = 2304, C4 = 3072, MM = Bz * Tt };

#if __has_builtin(__builtin_amdgcn_tensor_load_to_lds) && __has_builtin(__builtin_amdgcn_s_wait_tensorcnt)
#define USE_TDM 1
#else
#define USE_TDM 0
#endif

__device__ inline v8bf ld8(const bf16* p) { return *(const v8bf*)p; }
__device__ inline v16bf ld16(const bf16* p) { return *(const v16bf*)p; }
__device__ inline v16bf cat8(v8bf lo, v8bf hi) {
  return __builtin_shufflevector(lo, hi, 0,1,2,3,4,5,6,7,8,9,10,11,12,13,14,15);
}
__device__ inline v8f wmma_bf(v16bf a, v16bf b, v8f c) {
  // D(f32 16x16) = A(bf16 16x32) * B(bf16 32x16) + C
  return __builtin_amdgcn_wmma_f32_16x16x32_bf16(false, a, false, b, (short)0, c, false, false);
}
__device__ inline v8f zero8() {
  v8f z;
  #pragma unroll
  for (int e = 0; e < 8; ++e) z[e] = 0.0f;
  return z;
}
__device__ inline unsigned ldsoff(const void* p) {
  // AMDGPU generic pointer to LDS: low 32 bits are the workgroup LDS byte offset
  return (unsigned)(size_t)p;
}

#if USE_TDM
// Tensor DMA: 2D tile (tile_k contiguous elems x tile_rows rows) of a bf16
// row-major tensor [tensor_rows x tensor_k], row stride = stride_elems.
// D# packing per CDNA5 ISA.
// This toolchain's builtin takes 6 args: (g0, g1, g2, g3, g4, cpol).
__device__ inline void tdm_load_2d(unsigned lds_byte_off, const void* gaddr,
                                   unsigned tile_k, unsigned tile_rows,
                                   unsigned tensor_k, unsigned tensor_rows,
                                   unsigned stride_elems) {
  unsigned long long ga = (unsigned long long)(size_t)gaddr;
  v4u g0;
  g0[0] = 1u;                                             // count=1 (valid), user mode
  g0[1] = lds_byte_off;                                   // lds_addr [63:32]
  g0[2] = (unsigned)(ga & 0xffffffffu);                   // global_addr [95:64]
  g0[3] = (unsigned)((ga >> 32) & 0x1ffffffu)             // global_addr [120:96]
        | (2u << 30);                                     // type=2 ("image") [127:126]
  v8i_ g1;
  g1[0] = (int)(1u << 16);                                // wg_mask=0, data_size=1 (2B)
  g1[1] = (int)((tensor_k & 0xffffu) << 16);              // atomic_addr=0 | tensor_dim0 lo
  g1[2] = (int)(((tensor_k >> 16) & 0xffffu) | ((tensor_rows & 0xffffu) << 16));
  g1[3] = (int)(((tensor_rows >> 16) & 0xffffu) | ((tile_k & 0xffffu) << 16)); // tile_dim0
  g1[4] = (int)(tile_rows & 0xffffu);                     // tile_dim1 | tile_dim2=0
  g1[5] = (int)stride_elems;                              // tensor_dim0_stride [31:0]
  g1[6] = 0;                                              // stride hi16=0 | dim1_stride lo=0
  g1[7] = 0;
  v4i_ gz;
  gz[0] = 0; gz[1] = 0; gz[2] = 0; gz[3] = 0;             // groups 2/3: 2D tile, no iterate
  v8i_ gz8;
  #pragma unroll
  for (int e = 0; e < 8; ++e) gz8[e] = 0;
  __builtin_amdgcn_tensor_load_to_lds(g0, g1, gz, gz, gz8, 0);
}
#endif

// ---------------------------------------------------------------------------
// fp32 -> bf16 convert with transpose: wt[n*K + k] = (bf16) w[k*N + n]
// ---------------------------------------------------------------------------
__global__ void conv_t_kernel(const float* __restrict__ w, bf16* __restrict__ wt,
                              int K, int N) {
  long long idx = (long long)blockIdx.x * 256 + threadIdx.x;
  if (idx >= (long long)K * N) return;
  int n = (int)(idx / K);
  int k = (int)(idx % K);
  wt[idx] = (bf16)w[(long long)k * N + n];
}

// ---------------------------------------------------------------------------
// LayerNorm over C=768, one row per 256-thread block, bf16 output
// ---------------------------------------------------------------------------
__global__ __launch_bounds__(256) void ln_kernel(const float* __restrict__ x,
                                                 const float* __restrict__ sc,
                                                 const float* __restrict__ bi,
                                                 bf16* __restrict__ out) {
  __shared__ float s_sum[256], s_sq[256];
  const int row = blockIdx.x;
  const float* xr = x + (long long)row * Cc;
  float sum = 0.f, sq = 0.f;
  for (int i = threadIdx.x; i < Cc; i += 256) { float v = xr[i]; sum += v; sq += v * v; }
  s_sum[threadIdx.x] = sum; s_sq[threadIdx.x] = sq;
  __syncthreads();
  for (int st = 128; st > 0; st >>= 1) {
    if ((int)threadIdx.x < st) {
      s_sum[threadIdx.x] += s_sum[threadIdx.x + st];
      s_sq[threadIdx.x]  += s_sq[threadIdx.x + st];
    }
    __syncthreads();
  }
  const float mean = s_sum[0] * (1.0f / Cc);
  const float var  = s_sq[0]  * (1.0f / Cc) - mean * mean;
  const float rstd = rsqrtf(var + 1e-5f);
  bf16* orow = out + (long long)row * Cc;
  for (int i = threadIdx.x; i < Cc; i += 256)
    orow[i] = (bf16)((xr[i] - mean) * rstd * sc[i] + bi[i]);
}

// ---------------------------------------------------------------------------
// WMMA GEMM: out[M,N] = A[M,K](bf16) @ Wt[N,K]^T(bf16) + bias, epilogue variants
// Block: 256 thr = 8 waves (4 along M x 2 along N) -> 128x128 tile, K-step 32.
// TDM path: double-buffered LDS staging, one tensor_load_to_lds pair per step
// issued by wave 0, synchronized with s_wait_tensorcnt + barrier.
// EPI 0: bias -> bf16 | EPI 1: bias + resid -> fp32 | EPI 2: bias + gelu -> bf16
// ---------------------------------------------------------------------------
template <int EPI>
__global__ __launch_bounds__(256) void gemm_kernel(
    const bf16* __restrict__ A, const bf16* __restrict__ Wt,
    const float* __restrict__ bias, const float* __restrict__ resid,
    float* __restrict__ outF, bf16* __restrict__ outB, int M, int N, int K) {
  const int lane = threadIdx.x & 31;
  const int w    = threadIdx.x >> 5;
  const int wm = w & 3, wn = w >> 2;
  const int m0 = blockIdx.y * 128 + wm * 32;
  const int n0 = blockIdx.x * 128 + wn * 64;
  const int l16 = lane & 15;
  const int lhi = lane >> 4;  // 0|1

  v8f acc[2][4];
  #pragma unroll
  for (int i = 0; i < 2; ++i)
    #pragma unroll
    for (int j = 0; j < 4; ++j) acc[i][j] = zero8();

#if USE_TDM
  __shared__ __align__(128) bf16 sA[2][128 * 32];   // [row][k] per buffer
  __shared__ __align__(128) bf16 sB[2][128 * 32];   // [n-row][k] per buffer
  const long long arow_blk = (long long)blockIdx.y * 128 * K;
  const long long brow_blk = (long long)blockIdx.x * 128 * K;
  const int nk = K >> 5;

  if (threadIdx.x < 32) {
    tdm_load_2d(ldsoff(&sA[0][0]), A + arow_blk, 32, 128, K, M, K);
    tdm_load_2d(ldsoff(&sB[0][0]), Wt + brow_blk, 32, 128, K, N, K);
  }
  for (int kk = 0; kk < nk; ++kk) {
    const int buf = kk & 1;
    if (threadIdx.x < 32) {
      if (kk + 1 < nk) {
        const int k0n = (kk + 1) << 5;
        tdm_load_2d(ldsoff(&sA[buf ^ 1][0]), A + arow_blk + k0n, 32, 128, K, M, K);
        tdm_load_2d(ldsoff(&sB[buf ^ 1][0]), Wt + brow_blk + k0n, 32, 128, K, N, K);
        __builtin_amdgcn_s_wait_tensorcnt(2);   // current buffer's (older) DMAs done
      } else {
        __builtin_amdgcn_s_wait_tensorcnt(0);
      }
    }
    __syncthreads();
    v16bf af[2];
    {
      const bf16* p0 = &sA[buf][(wm * 32 + l16) * 32 + lhi * 8];
      af[0] = cat8(ld8(p0), ld8(p0 + 16));
      const bf16* p1 = &sA[buf][(wm * 32 + 16 + l16) * 32 + lhi * 8];
      af[1] = cat8(ld8(p1), ld8(p1 + 16));
    }
    v16bf bfr[4];
    #pragma unroll
    for (int j = 0; j < 4; ++j)
      bfr[j] = cat8(ld8(&sB[buf][(wn * 64 + j * 16 + l16) * 32 + lhi * 8]), ld8(&sB[buf][(wn * 64 + j * 16 + l16) * 32 + lhi * 8 + 16]));
    #pragma unroll
    for (int i = 0; i < 2; ++i)
      #pragma unroll
      for (int j = 0; j < 4; ++j) acc[i][j] = wmma_bf(af[i], bfr[j], acc[i][j]);
    __syncthreads();   // all reads of buf done before it is re-filled
  }
#else
  const long long arow0 = (long long)(m0 + l16) * K;
  const long long arow1 = (long long)(m0 + 16 + l16) * K;
  for (int k0 = 0; k0 < K; k0 += 32) {
    if (k0 + 32 < K) {
      __builtin_prefetch(A + arow0 + k0 + 32, 0, 1);
      __builtin_prefetch(A + arow1 + k0 + 32, 0, 1);
    }
    v16bf af[2];
    {
      const bf16* p0 = A + arow0 + k0 + lhi * 8;
      af[0] = cat8(ld8(p0), ld8(p0 + 16));
      const bf16* p1 = A + arow1 + k0 + lhi * 8;
      af[1] = cat8(ld8(p1), ld8(p1 + 16));
    }
    v16bf bfr[4];
    #pragma unroll
    for (int j = 0; j < 4; ++j)
      bfr[j] = cat8(ld8(Wt + (long long)(n0 + j * 16 + l16) * K + k0 + lhi * 8), ld8(Wt + (long long)(n0 + j * 16 + l16) * K + k0 + lhi * 8 + 16));
    #pragma unroll
    for (int i = 0; i < 2; ++i)
      #pragma unroll
      for (int j = 0; j < 4; ++j) acc[i][j] = wmma_bf(af[i], bfr[j], acc[i][j]);
  }
#endif

  #pragma unroll
  for (int i = 0; i < 2; ++i) {
    #pragma unroll
    for (int j = 0; j < 4; ++j) {
      const int n = n0 + j * 16 + l16;
      const float bv = bias[n];
      #pragma unroll
      for (int r = 0; r < 8; ++r) {
        const int m = m0 + i * 16 + r + lhi * 8;       // C/D layout: half-wave rows r / 8+r
        const long long off = (long long)m * N + n;
        float v = acc[i][j][r] + bv;
        if (EPI == 0) {
          outB[off] = (bf16)v;
        } else if (EPI == 1) {
          outF[off] = v + resid[off];
        } else {
          float g = 0.5f * v * (1.0f + tanhf(0.7978845608028654f * (v + 0.044715f * v * v * v)));
          outB[off] = (bf16)g;
        }
      }
    }
  }
}

// ---------------------------------------------------------------------------
// Causal flash attention. qkv[B,T,3C] bf16 (q|k|v each head-major HD=64).
// Block: 128 thr = 4 waves; wave owns 16 q rows; grid (T/64, B*H).
// S = Q K^T via WMMA (K-dim = HD, 2 steps), online softmax, ctx += P V.
// ---------------------------------------------------------------------------
__global__ __launch_bounds__(128) void attn_kernel(const bf16* __restrict__ qkv,
                                                   bf16* __restrict__ ctx) {
  __shared__ __align__(64) bf16 lds_vt[4][HDd * 32];  // V^T per wave: [d][token]
  __shared__ __align__(64) bf16 lds_p[4][16 * 32];    // P per wave:  [qrow][k]
  const int lane = threadIdx.x & 31;
  const int w    = threadIdx.x >> 5;
  const int l16 = lane & 15;
  const int lhi = lane >> 4;
  const int b = blockIdx.y / Hh, h = blockIdx.y % Hh;
  const int q0 = blockIdx.x * 64 + w * 16;
  const long long bt = (long long)b * Tt;

  // Q A-fragments (held for whole kernel): d-chunks {0..31},{32..63}
  v16bf qf[2];
  {
    const bf16* p = qkv + (bt + q0 + l16) * C3 + h * HDd + lhi * 8;
    qf[0] = cat8(ld8(p),      ld8(p + 16));
    qf[1] = cat8(ld8(p + 32), ld8(p + 48));
  }

  float mrow[8], lrow[8];
  v8f cacc[4];
  #pragma unroll
  for (int r = 0; r < 8; ++r) { mrow[r] = -1e30f; lrow[r] = 0.0f; }
  #pragma unroll
  for (int dt = 0; dt < 4; ++dt) cacc[dt] = zero8();

  const int kend = q0 + 16;                     // causal: keys <= q0w+15
  for (int kt = 0; kt < kend; kt += 32) {
    // ---- stage V tile (32 tokens x 64 d) transposed into wave-private LDS
    {
      int tok = kt + lane; if (tok > Tt - 1) tok = Tt - 1;
      const bf16* vp = qkv + (bt + tok) * C3 + 2 * Cc + h * HDd;
      #pragma unroll
      for (int e8 = 0; e8 < 8; ++e8) {
        v8bf vv = ld8(vp + e8 * 8);
        #pragma unroll
        for (int e = 0; e < 8; ++e) lds_vt[w][(e8 * 8 + e) * 32 + lane] = vv[e];
      }
    }
    // ---- S = Q K^T for two 16-key subtiles (B frag = contiguous read of K rows)
    v8f sacc[2];
    #pragma unroll
    for (int ks = 0; ks < 2; ++ks) {
      int tok = kt + ks * 16 + l16; if (tok > Tt - 1) tok = Tt - 1;
      const bf16* kp = qkv + (bt + tok) * C3 + Cc + h * HDd + lhi * 8;
      v8f s = zero8();
      s = wmma_bf(qf[0], cat8(ld8(kp), ld8(kp + 16)),      s);
      s = wmma_bf(qf[1], cat8(ld8(kp + 32), ld8(kp + 48)), s);
      sacc[ks] = s;
    }
    // ---- scale + causal mask
    #pragma unroll
    for (int ks = 0; ks < 2; ++ks) {
      const int kcol = kt + ks * 16 + l16;
      #pragma unroll
      for (int r = 0; r < 8; ++r) {
        const int qrow = q0 + r + lhi * 8;
        sacc[ks][r] = (kcol <= qrow) ? sacc[ks][r] * 0.125f : -1e30f;
      }
    }
    // ---- online softmax (row reductions within each 16-lane half)
    #pragma unroll
    for (int r = 0; r < 8; ++r) {
      float tmax = fmaxf(sacc[0][r], sacc[1][r]);
      #pragma unroll
      for (int sh = 8; sh >= 1; sh >>= 1) tmax = fmaxf(tmax, __shfl_xor(tmax, sh, 32));
      const float mnew  = fmaxf(mrow[r], tmax);
      const float alpha = __expf(mrow[r] - mnew);
      mrow[r] = mnew;
      const float p0 = __expf(sacc[0][r] - mnew);
      const float p1 = __expf(sacc[1][r] - mnew);
      sacc[0][r] = p0; sacc[1][r] = p1;
      float rsum = p0 + p1;
      #pragma unroll
      for (int sh = 8; sh >= 1; sh >>= 1) rsum += __shfl_xor(rsum, sh, 32);
      lrow[r] = lrow[r] * alpha + rsum;
      #pragma unroll
      for (int dt = 0; dt < 4; ++dt) cacc[dt][r] *= alpha;
    }
    // ---- P (D layout) -> LDS -> A-fragment layout
    #pragma unroll
    for (int ks = 0; ks < 2; ++ks)
      #pragma unroll
      for (int r = 0; r < 8; ++r)
        lds_p[w][(r + lhi * 8) * 32 + ks * 16 + l16] = (bf16)sacc[ks][r];
    const bf16* pp = &lds_p[w][l16 * 32 + lhi * 8];
    v16bf pf = cat8(ld8(pp), ld8(pp + 16));
    // ---- ctx += P V  (B frag from transposed V in LDS)
    #pragma unroll
    for (int dt = 0; dt < 4; ++dt) {
      const bf16* vb = &lds_vt[w][(dt * 16 + l16) * 32 + lhi * 8];
      cacc[dt] = wmma_bf(pf, cat8(ld8(vb), ld8(vb + 16)), cacc[dt]);
    }
  }
  // ---- normalize + write ctx (bf16, [B,T,C] head-major)
  #pragma unroll
  for (int dt = 0; dt < 4; ++dt)
    #pragma unroll
    for (int r = 0; r < 8; ++r) {
      const int q = q0 + r + lhi * 8;
      ctx[(bt + q) * Cc + h * HDd + dt * 16 + l16] = (bf16)(cacc[dt][r] / lrow[r]);
    }
}

// ---------------------------------------------------------------------------
extern "C" void kernel_launch(void* const* d_in, const int* in_sizes, int n_in,
                              void* d_out, int out_size, void* d_ws, size_t ws_size,
                              hipStream_t stream) {
  const float* x      = (const float*)d_in[0];
  // d_in[1] = mask (causal, implicit)
  const float* ln1_s  = (const float*)d_in[2];
  const float* ln1_b  = (const float*)d_in[3];
  const float* w_attn = (const float*)d_in[4];
  const float* b_attn = (const float*)d_in[5];
  const float* w_o    = (const float*)d_in[6];
  const float* b_o    = (const float*)d_in[7];
  const float* ln2_s  = (const float*)d_in[8];
  const float* ln2_b  = (const float*)d_in[9];
  const float* w_fc   = (const float*)d_in[10];
  const float* b_fc   = (const float*)d_in[11];
  const float* w_proj = (const float*)d_in[12];
  const float* b_proj = (const float*)d_in[13];
  float* out = (float*)d_out;

  char* ws = (char*)d_ws;
  size_t off = 0;
  auto alloc = [&](size_t bytes) { void* p = ws + off; off += (bytes + 255) & ~(size_t)255; return p; };
  bf16* w_attn_t = (bf16*)alloc((size_t)Cc * C3 * 2);
  bf16* w_o_t    = (bf16*)alloc((size_t)Cc * Cc * 2);
  bf16* w_fc_t   = (bf16*)alloc((size_t)Cc * C4 * 2);
  bf16* w_proj_t = (bf16*)alloc((size_t)C4 * Cc * 2);
  bf16* hbuf     = (bf16*)alloc((size_t)MM * Cc * 2);  // h1 -> ctx -> h2 (sequential reuse)
  bf16* bigbuf   = (bf16*)alloc((size_t)MM * C4 * 2);  // qkv (stride 2304) then fc (stride 3072)
  float* x1      = (float*)alloc((size_t)MM * Cc * 4);

  auto cdiv = [](long long a, long long b) { return (unsigned)((a + b - 1) / b); };
  dim3 blk256(256);

  conv_t_kernel<<<cdiv((long long)Cc * C3, 256), blk256, 0, stream>>>(w_attn, w_attn_t, Cc, C3);
  conv_t_kernel<<<cdiv((long long)Cc * Cc, 256), blk256, 0, stream>>>(w_o,    w_o_t,    Cc, Cc);
  conv_t_kernel<<<cdiv((long long)Cc * C4, 256), blk256, 0, stream>>>(w_fc,   w_fc_t,   Cc, C4);
  conv_t_kernel<<<cdiv((long long)C4 * Cc, 256), blk256, 0, stream>>>(w_proj, w_proj_t, C4, Cc);

  // attention branch
  ln_kernel<<<MM, 256, 0, stream>>>(x, ln1_s, ln1_b, hbuf);
  gemm_kernel<0><<<dim3(C3 / 128, MM / 128), 256, 0, stream>>>(
      hbuf, w_attn_t, b_attn, nullptr, nullptr, bigbuf, MM, C3, Cc);
  attn_kernel<<<dim3(Tt / 64, Bz * Hh), 128, 0, stream>>>(bigbuf, hbuf);
  gemm_kernel<1><<<dim3(Cc / 128, MM / 128), 256, 0, stream>>>(
      hbuf, w_o_t, b_o, x, x1, nullptr, MM, Cc, Cc);

  // MLP branch
  ln_kernel<<<MM, 256, 0, stream>>>(x1, ln2_s, ln2_b, hbuf);
  gemm_kernel<2><<<dim3(C4 / 128, MM / 128), 256, 0, stream>>>(
      hbuf, w_fc_t, b_fc, nullptr, nullptr, bigbuf, MM, C4, Cc);
  gemm_kernel<1><<<dim3(Cc / 128, MM / 128), 256, 0, stream>>>(
      bigbuf, w_proj_t, b_proj, x1, out, nullptr, MM, Cc, C4);
}
